// TA_LSTM_1202590843337
// MI455X (gfx1250) — compile-verified, matrix operand order checked by reading, not run on hardware
//
#include <hip/hip_runtime.h>
#include <hip/hip_bf16.h>
#include <stdint.h>

// ---------------------------------------------------------------------------
// TA-LSTM fused kernel for gfx1250 (MI455X).
// T=50, B=4096, IN_DIM=64, H=128, gates 4H=512.
// Per-wave persistent 16-row batch tile; v_wmma_f32_16x16x32_f16 for the
// recurrent GEMMs and the attention GEMM; v_tanh_f32 for activations.
// ---------------------------------------------------------------------------

typedef __attribute__((ext_vector_type(16))) _Float16 v16h;
typedef __attribute__((ext_vector_type(8)))  _Float16 v8h;
typedef __attribute__((ext_vector_type(4)))  _Float16 v4h;
typedef __attribute__((ext_vector_type(8)))  float    v8f;

#define TT   50
#define BB   4096
#define IND  64
#define HD   128
#define G4   512

// ---- WMMA operand loaders -------------------------------------------------
// 16-bit A-matrix 16x32 layout (ISA 7.12.2): lane l holds row m = l&15;
// lanes 0-15 hold K = {0..7, 16..23}, lanes 16-31 hold K = {8..15, 24..31}.
// B (32x16, K x N) uses the same packing on B^T, i.e. row n = l&15 of W.
__device__ __forceinline__ v16h ld_tile(const _Float16* base, int stride, int lane) {
    const int m  = lane & 15;
    const int kh = (lane >> 4) & 1;
    const _Float16* p = base + (size_t)m * stride + kh * 8;
    v8h lo = *(const v8h*)(p);
    v8h hi = *(const v8h*)(p + 16);
    v16h r;
#pragma unroll
    for (int i = 0; i < 8; ++i) { r[i] = lo[i]; r[i + 8] = hi[i]; }
    return r;
}

__device__ __forceinline__ v16h ld_tile_guard(const _Float16* base, int stride,
                                              int lane, int nrows) {
    const int m  = lane & 15;
    const int kh = (lane >> 4) & 1;
    v16h r;
    if (m >= nrows) {
#pragma unroll
        for (int i = 0; i < 16; ++i) r[i] = (_Float16)0.0f;
        return r;
    }
    const _Float16* p = base + (size_t)m * stride + kh * 8;
    v8h lo = *(const v8h*)(p);
    v8h hi = *(const v8h*)(p + 16);
#pragma unroll
    for (int i = 0; i < 8; ++i) { r[i] = lo[i]; r[i + 8] = hi[i]; }
    return r;
}

#define WMMA_F16(A, B, C) \
    __builtin_amdgcn_wmma_f32_16x16x32_f16(false, (A), false, (B), (short)0, (C), false, false)

// ---- fast activations (v_tanh_f32 on gfx1250, expf fallback) --------------
__device__ __forceinline__ float tanh_f(float x) {
#if __has_builtin(__builtin_amdgcn_tanhf)
    return __builtin_amdgcn_tanhf(x);
#else
    float a = __builtin_fabsf(x);
    float e = __expf(-2.0f * a);
    float t = (1.0f - e) * __builtin_amdgcn_rcpf(1.0f + e);
    return __builtin_copysignf(t, x);
#endif
}
__device__ __forceinline__ float sigmoid_f(float x) {
    // sigmoid(x) = 0.5*tanh(0.5x) + 0.5  -> v_mul, v_tanh, v_fma
    return __builtin_fmaf(0.5f, tanh_f(0.5f * x), 0.5f);
}

// ---- pre-pass kernels -----------------------------------------------------
__global__ void cvt_f32_to_f16(const float* __restrict__ src,
                               _Float16* __restrict__ dst, int n) {
    int i = blockIdx.x * blockDim.x + threadIdx.x;
    if (i < n) dst[i] = (_Float16)src[i];
}

__global__ void bias_sum_k(const float* __restrict__ a, const float* __restrict__ b,
                           float* __restrict__ dst, int n) {
    int i = blockIdx.x * blockDim.x + threadIdx.x;
    if (i < n) dst[i] = a[i] + b[i];
}

// ---- main fused kernel ----------------------------------------------------
// grid = 64 blocks, block = 128 threads (4 waves); each wave owns 16 batch rows.
// Dynamic LDS: sWih 64KB | sWhh 128KB | sH 4x(16x136 f16) | sS 4x(16x64 f32)
//            | sB 4x(16x52 f32)  => 243712 B < 320 KB WGP LDS.
__global__ __launch_bounds__(128) void ta_lstm_main(
    const _Float16* __restrict__ x16,
    const _Float16* __restrict__ Wih16,
    const _Float16* __restrict__ Whh16,
    const _Float16* __restrict__ Wta16,
    const float*    __restrict__ biasg,
    const float*    __restrict__ b_ta,
    _Float16*       __restrict__ hs,
    float*          __restrict__ out)
{
    extern __shared__ char smem[];
    _Float16* sWih  = (_Float16*)smem;                       // 512*64
    _Float16* sWhh  = sWih + G4 * IND;                       // 512*128
    _Float16* sHall = sWhh + G4 * HD;                        // 4*16*136
    float*    sSall = (float*)(sHall + 4 * 16 * 136);        // 4*16*64
    float*    sBall = sSall + 4 * 16 * 64;                   // 4*16*52

    const int tid  = threadIdx.x;
    const int wave = tid >> 5;
    const int lane = tid & 31;
    const int nsel = lane & 15;

    _Float16* sH = sHall + wave * (16 * 136);
    float*    sS = sSall + wave * (16 * 64);
    float*    sB = sBall + wave * (16 * 52);

    // cooperative weight load into LDS (u32 copies)
    {
        const uint32_t* sA = (const uint32_t*)Wih16;
        uint32_t*       dA = (uint32_t*)sWih;
        for (int i = tid; i < G4 * IND / 2; i += 128) dA[i] = sA[i];
        const uint32_t* sBw = (const uint32_t*)Whh16;
        uint32_t*       dBw = (uint32_t*)sWhh;
        for (int i = tid; i < G4 * HD / 2; i += 128) dBw[i] = sBw[i];
    }
    // zero per-wave h stage (h0 = 0)
    for (int i = lane; i < 16 * 136; i += 32) sH[i] = (_Float16)0.0f;
    __syncthreads();

    const int b0 = blockIdx.x * 64 + wave * 16;

    // t-invariant per-lane bias values: bj[j][g] = (b_ih+b_hh)[g*128 + j*16 + nsel]
    float bj[8][4];
#pragma unroll
    for (int j = 0; j < 8; ++j)
#pragma unroll
        for (int g = 0; g < 4; ++g) bj[j][g] = biasg[g * HD + j * 16 + nsel];

    // cell state, D-layout: element e of chunk j == (m = e or 8+e, n = j*16 + nsel)
    v8f cst[8];
#pragma unroll
    for (int j = 0; j < 8; ++j)
#pragma unroll
        for (int e = 0; e < 8; ++e) cst[j][e] = 0.0f;

    // ---------------- LSTM recurrence ----------------
    for (int t = 0; t < TT; ++t) {
        const _Float16* xrow = x16 + ((size_t)t * BB + b0) * IND;
        v16h Ax0 = ld_tile(xrow,      IND, lane);
        v16h Ax1 = ld_tile(xrow + 32, IND, lane);
        v16h Ah0 = ld_tile(sH,        136, lane);
        v16h Ah1 = ld_tile(sH + 32,   136, lane);
        v16h Ah2 = ld_tile(sH + 64,   136, lane);
        v16h Ah3 = ld_tile(sH + 96,   136, lane);

#pragma unroll
        for (int j = 0; j < 8; ++j) {
            v8f acc[4];
#pragma unroll
            for (int g = 0; g < 4; ++g) {
                const int n0 = g * HD + j * 16;
                v8f c0 = {};                       // inline-0 C on first WMMA
                v16h Bm;
                Bm = ld_tile(sWih + (size_t)n0 * IND,      IND, lane);
                c0 = WMMA_F16(Ax0, Bm, c0);
                Bm = ld_tile(sWih + (size_t)n0 * IND + 32, IND, lane);
                c0 = WMMA_F16(Ax1, Bm, c0);
                Bm = ld_tile(sWhh + (size_t)n0 * HD,       HD, lane);
                c0 = WMMA_F16(Ah0, Bm, c0);
                Bm = ld_tile(sWhh + (size_t)n0 * HD + 32,  HD, lane);
                c0 = WMMA_F16(Ah1, Bm, c0);
                Bm = ld_tile(sWhh + (size_t)n0 * HD + 64,  HD, lane);
                c0 = WMMA_F16(Ah2, Bm, c0);
                Bm = ld_tile(sWhh + (size_t)n0 * HD + 96,  HD, lane);
                c0 = WMMA_F16(Ah3, Bm, c0);
                acc[g] = c0;
            }
            // elementwise gate math; D layout (m,n) identical across the 4 tiles
#pragma unroll
            for (int e = 0; e < 8; ++e) {
                float iv = sigmoid_f(acc[0][e] + bj[j][0]);
                float fv = sigmoid_f(acc[1][e] + bj[j][1]);
                float gv = tanh_f(acc[2][e] + bj[j][2]);
                float ov = sigmoid_f(acc[3][e] + bj[j][3]);
                float cn = __builtin_fmaf(fv, cst[j][e], iv * gv);
                cst[j][e] = cn;
                float hv = ov * tanh_f(cn);
                const int m = (lane < 16) ? e : (8 + e);
                sH[m * 136 + j * 16 + nsel] = (_Float16)hv;
            }
        }
        // vectorized coalesced copy of this step's h tile: LDS -> global hs
        // (same-wave LDS ops are in-order; loads below see the stores above)
        {
            const int mrow  = nsel;          // 16 rows
            const int halfc = lane >> 4;     // 2 column halves of 64
            const _Float16* srcp = sH + mrow * 136 + halfc * 64;
            _Float16* dstp = hs + ((size_t)t * BB + b0 + mrow) * HD + halfc * 64;
#pragma unroll
            for (int q = 0; q < 8; ++q) {
                v8h v = *(const v8h*)(srcp + q * 8);
                *(v8h*)(dstp + q * 8) = v;
            }
        }
    }

    // ---------------- attention logits: [16 x 50] = total_ht @ W_ta^T ------
    v8f lg[4];
#pragma unroll
    for (int tb = 0; tb < 4; ++tb) {
        const int tau = tb * 16 + nsel;
        const float bv = (tau < TT) ? b_ta[tau] : 0.0f;
#pragma unroll
        for (int e = 0; e < 8; ++e) lg[tb][e] = bv;
    }
    for (int kc = 0; kc < (TT * HD) / 32; ++kc) {     // 200 K-chunks
        const int tt = kc >> 2;
        const int h0 = (kc & 3) * 32;
        v16h Aa = ld_tile(hs + ((size_t)tt * BB + b0) * HD + h0, HD, lane);
#pragma unroll
        for (int tb = 0; tb < 4; ++tb) {
            const int rows = (tb * 16 + 16 <= TT) ? 16 : (TT - tb * 16);
            v16h Bm = ld_tile_guard(Wta16 + (size_t)tb * 16 * (TT * HD) + kc * 32,
                                    TT * HD, lane, rows);
            lg[tb] = WMMA_F16(Aa, Bm, lg[tb]);
        }
    }
    // dump logits to LDS [16][64]
#pragma unroll
    for (int tb = 0; tb < 4; ++tb)
#pragma unroll
        for (int e = 0; e < 8; ++e) {
            const int m = (lane < 16) ? e : (8 + e);
            sS[m * 64 + tb * 16 + nsel] = lg[tb][e];
        }
    asm volatile("s_wait_dscnt 0" ::: "memory");   // cross-lane LDS exchange

    // relu -> softmax over 50, one row per lane (lanes 0..15)
    if (lane < 16) {
        const int m = lane;
        float r[TT];
        float mx = 0.0f;
#pragma unroll 1
        for (int i = 0; i < TT; ++i) {
            float v = sS[m * 64 + i];
            v = v > 0.0f ? v : 0.0f;
            r[i] = v;
            mx = v > mx ? v : mx;
        }
        float sum = 0.0f;
#pragma unroll 1
        for (int i = 0; i < TT; ++i) { float e = __expf(r[i] - mx); r[i] = e; sum += e; }
        const float inv = __builtin_amdgcn_rcpf(sum);
#pragma unroll 1
        for (int i = 0; i < TT; ++i) sB[m * 52 + i] = r[i] * inv;
    }
    asm volatile("s_wait_dscnt 0" ::: "memory");   // cross-lane LDS exchange

    // ---------------- out[b,:] = sum_t beta[b,t] * hs[t,b,:] ---------------
    {
        const int c0 = lane * 4;                      // 32 lanes x 4 cols = 128
        for (int m = 0; m < 16; ++m) {
            float a0 = 0.f, a1 = 0.f, a2 = 0.f, a3 = 0.f;
#pragma unroll 1
            for (int t = 0; t < TT; ++t) {
                const float bt = sB[m * 52 + t];
                v4h hv = *(const v4h*)(hs + ((size_t)t * BB + b0 + m) * HD + c0);
                a0 += bt * (float)hv[0];
                a1 += bt * (float)hv[1];
                a2 += bt * (float)hv[2];
                a3 += bt * (float)hv[3];
            }
            float* orow = out + (size_t)(b0 + m) * HD + c0;
            orow[0] = a0; orow[1] = a1; orow[2] = a2; orow[3] = a3;
        }
    }
}

// ---------------------------------------------------------------------------
extern "C" void kernel_launch(void* const* d_in, const int* in_sizes, int n_in,
                              void* d_out, int out_size, void* d_ws, size_t ws_size,
                              hipStream_t stream) {
    (void)in_sizes; (void)n_in; (void)out_size; (void)ws_size;

    const float* input = (const float*)d_in[0];   // [T,B,64]
    const float* W_ih  = (const float*)d_in[1];   // [512,64]
    const float* W_hh  = (const float*)d_in[2];   // [512,128]
    const float* b_ih  = (const float*)d_in[3];   // [512]
    const float* b_hh  = (const float*)d_in[4];   // [512]
    const float* W_ta  = (const float*)d_in[5];   // [50,6400]
    const float* b_ta  = (const float*)d_in[6];   // [50]
    float* out = (float*)d_out;                   // [B,128]

    // workspace layout (bytes)
    char* w = (char*)d_ws;
    const size_t HS_B   = (size_t)TT * BB * HD * 2;      // 52,428,800
    const size_t X_B    = (size_t)TT * BB * IND * 2;     // 26,214,400
    const size_t WIH_B  = (size_t)G4 * IND * 2;          // 65,536
    const size_t WHH_B  = (size_t)G4 * HD * 2;           // 131,072
    const size_t WTA_B  = (size_t)TT * TT * HD * 2;      // 640,000
    _Float16* hs    = (_Float16*)(w);
    _Float16* x16   = (_Float16*)(w + HS_B);
    _Float16* Wih16 = (_Float16*)(w + HS_B + X_B);
    _Float16* Whh16 = (_Float16*)(w + HS_B + X_B + WIH_B);
    _Float16* Wta16 = (_Float16*)(w + HS_B + X_B + WIH_B + WHH_B);
    float*    biasg = (float*)   (w + HS_B + X_B + WIH_B + WHH_B + WTA_B);

    // pre-pass: fp32 -> fp16 conversions + fused bias
    {
        int n;
        n = TT * BB * IND;
        cvt_f32_to_f16<<<(n + 255) / 256, 256, 0, stream>>>(input, x16, n);
        n = G4 * IND;
        cvt_f32_to_f16<<<(n + 255) / 256, 256, 0, stream>>>(W_ih, Wih16, n);
        n = G4 * HD;
        cvt_f32_to_f16<<<(n + 255) / 256, 256, 0, stream>>>(W_hh, Whh16, n);
        n = TT * TT * HD;
        cvt_f32_to_f16<<<(n + 255) / 256, 256, 0, stream>>>(W_ta, Wta16, n);
        n = G4;
        bias_sum_k<<<(n + 255) / 256, 256, 0, stream>>>(b_ih, b_hh, biasg, n);
    }

    // fused LSTM + attention: 64 blocks x 128 threads, 243712 B dynamic LDS
    const size_t smem = (size_t)(G4 * IND + G4 * HD + 4 * 16 * 136) * 2
                      + (size_t)(4 * 16 * 64 + 4 * 16 * 52) * 4;
    ta_lstm_main<<<dim3(BB / 64), dim3(128), smem, stream>>>(
        x16, Wih16, Whh16, Wta16, biasg, b_ta, hs, out);
}
